// EMAGateModel_56109452755474
// MI455X (gfx1250) — matrix operand order repaired and checked by reading, not compiled, not run on hardware
//
#include <hip/hip_runtime.h>
#include <hip/hip_bf16.h>
#include <cstdint>
#include <cstddef>

// ---------------- problem constants (reference) ----------------
#define BB 64
#define LL 1024
#define HH 256
#define H2 512
#define VV 32000
#define DECAY 0.05f          // 1 - ALPHA
#define GT2 0.16f            // GATE_THRESH^2
#define LN_EPS 1e-5f
#define NORM_EPS 1e-12f

typedef __bf16 bhalf;
typedef __attribute__((ext_vector_type(16))) __bf16 bhalf16;
typedef __attribute__((ext_vector_type(8)))  __bf16 bhalf8;
typedef __attribute__((ext_vector_type(8)))  float  floatx8;

// CDNA5 async global->LDS copy path (ASYNCcnt), guarded so the file always compiles.
#if defined(__has_builtin)
#if __has_builtin(__builtin_amdgcn_global_load_async_to_lds_b32) && \
    __has_builtin(__builtin_amdgcn_s_wait_asynccnt)
#define USE_ASYNC_LDS 1
#endif
#endif
#ifndef USE_ASYNC_LDS
#define USE_ASYNC_LDS 0
#endif

// builtin expects address-space-qualified int* operands
typedef __attribute__((address_space(1))) int as1i;  // global
typedef __attribute__((address_space(3))) int as3i;  // LDS

// ---------------- WMMA operand loaders ----------------
// A operand (16x32 bf16 tile, row-major source, per ISA layout:
// lane group g holds K = {8g..8g+7} U {16+8g..16+8g+7} of row (lane&15))
__device__ __forceinline__ bhalf16 load_a16(const bhalf* tile, int stride, int lane) {
  const int m = lane & 15, g = lane >> 4;
  const bhalf* r = tile + (size_t)m * stride + 8 * g;
  union { bhalf16 v; bhalf8 h[2]; } u;
  u.h[0] = *(const bhalf8*)(r);
  u.h[1] = *(const bhalf8*)(r + 16);
  return u.v;
}

// B operand (32x16 bf16 tile) from a PRE-TRANSPOSED weight Wt[N_total][K_total]:
// lane (n = lane&15, g = lane>>4) holds K = {8g..8g+7} U {16+8g..16+8g+7} of column n0+n,
// which is two 16-byte loads from Wt.
__device__ __forceinline__ bhalf16 load_bT16(const bhalf* wT, int ktot, int n0, int k0, int lane) {
  const int n = lane & 15, g = lane >> 4;
  const bhalf* r = wT + (size_t)(n0 + n) * ktot + k0 + 8 * g;
  union { bhalf16 v; bhalf8 h[2]; } u;
  u.h[0] = *(const bhalf8*)(r);
  u.h[1] = *(const bhalf8*)(r + 16);
  return u.v;
}

// ---------------- weight convert + transpose ----------------
__global__ void __launch_bounds__(256)
k_cvtT_bf16(const float* __restrict__ src, bhalf* __restrict__ dstT, int K, int N) {
  size_t idx = (size_t)blockIdx.x * 256 + threadIdx.x;
  if (idx >= (size_t)K * N) return;
  int k = (int)(idx / N), n = (int)(idx % N);
  dstT[(size_t)n * K + k] = (bhalf)src[idx];
}

__global__ void __launch_bounds__(256)
k_cvtT_f32(const float* __restrict__ src, float* __restrict__ dstT, int K, int N) {
  size_t idx = (size_t)blockIdx.x * 256 + threadIdx.x;
  if (idx >= (size_t)K * N) return;
  int k = (int)(idx / N), n = (int)(idx % N);
  dstT[(size_t)n * K + k] = src[idx];
}

// ---------------- fused embed gather + FFN + residual + LayerNorm ----------------
// one wave32 per 16-token tile
__global__ void __launch_bounds__(32)
k_embed_ffn_ln(const int* __restrict__ seq, const float* __restrict__ embed,
               const bhalf* __restrict__ W1T, const float* __restrict__ b1,
               const bhalf* __restrict__ W2T, const float* __restrict__ b2,
               const float* __restrict__ ln_g, const float* __restrict__ ln_b,
               bhalf* __restrict__ hbf) {
  __shared__ bhalf hA[16 * HH];     // bf16 A operand for GEMM1
  __shared__ float hres[16 * HH];   // fp32 residual / x buffer
  __shared__ bhalf uu[16 * H2];     // ReLU(h@W1+b1) as bf16 A operand for GEMM2
  __shared__ float mu_s[16], rs_s[16];

  const int lane = threadIdx.x;
  const size_t tok0 = (size_t)blockIdx.x * 16;

  // gather 16 embedding rows
  for (int idx = lane; idx < 16 * HH; idx += 32) {
    int r = idx >> 8, c = idx & (HH - 1);
    int tok = seq[tok0 + r];
    float v = embed[(size_t)tok * HH + c];
    hres[idx] = v;
    hA[idx] = (bhalf)v;
  }
  __syncthreads();

  const int n = lane & 15, g = lane >> 4;

  // GEMM1: u = ReLU(h @ W1 + b1)   [16,256]x[256,512]
  for (int nt = 0; nt < H2 / 16; ++nt) {
    floatx8 acc;
    float bias = b1[nt * 16 + n];
#pragma unroll
    for (int r = 0; r < 8; ++r) acc[r] = bias;
#pragma unroll
    for (int kt = 0; kt < HH / 32; ++kt) {
      bhalf16 a = load_a16(hA + kt * 32, HH, lane);
      bhalf16 b = load_bT16(W1T, HH, nt * 16, kt * 32, lane);
      acc = __builtin_amdgcn_wmma_f32_16x16x32_bf16(false, a, false, b, (short)0, acc, false, false);
    }
#pragma unroll
    for (int r = 0; r < 8; ++r) {
      float v = fmaxf(acc[r], 0.0f);
      uu[(r + 8 * g) * H2 + nt * 16 + n] = (bhalf)v;
    }
  }
  __syncthreads();

  // GEMM2: x = u @ W2 + b2 + h     [16,512]x[512,256]
  for (int nt = 0; nt < HH / 16; ++nt) {
    floatx8 acc;
    float bias = b2[nt * 16 + n];
#pragma unroll
    for (int r = 0; r < 8; ++r) acc[r] = bias;
#pragma unroll
    for (int kt = 0; kt < H2 / 32; ++kt) {
      bhalf16 a = load_a16(uu + kt * 32, H2, lane);
      bhalf16 b = load_bT16(W2T, H2, nt * 16, kt * 32, lane);
      acc = __builtin_amdgcn_wmma_f32_16x16x32_bf16(false, a, false, b, (short)0, acc, false, false);
    }
#pragma unroll
    for (int r = 0; r < 8; ++r) {
      int mrow = r + 8 * g, c = nt * 16 + n;
      hres[mrow * HH + c] += acc[r];   // residual add (unique element per lane/r)
    }
  }
  __syncthreads();

  // LayerNorm per row (population var)
  if (lane < 16) {
    const float* row = hres + lane * HH;
    float mu = 0.0f, sq = 0.0f;
    for (int c = 0; c < HH; ++c) { float v = row[c]; mu += v; sq += v * v; }
    mu *= (1.0f / HH);
    float var = sq * (1.0f / HH) - mu * mu;
    mu_s[lane] = mu;
    rs_s[lane] = rsqrtf(var + LN_EPS);
  }
  __syncthreads();
  for (int idx = lane; idx < 16 * HH; idx += 32) {
    int r = idx >> 8, c = idx & (HH - 1);
    float v = (hres[idx] - mu_s[r]) * rs_s[r] * ln_g[c] + ln_b[c];
    hbf[tok0 * HH + idx] = (bhalf)v;
  }
}

// ---------------- keys: k = h @ Wk (all L rows; last row doubles as q) ----------------
__global__ void __launch_bounds__(32)
k_keys(const bhalf* __restrict__ hbf, const bhalf* __restrict__ WkT, float* __restrict__ kout) {
  const int lane = threadIdx.x;
  const int n = lane & 15, g = lane >> 4;
  const size_t row0 = (size_t)blockIdx.x * 16;
  const bhalf* A = hbf + row0 * HH;
  for (int nt = 0; nt < HH / 16; ++nt) {
    floatx8 acc;
#pragma unroll
    for (int r = 0; r < 8; ++r) acc[r] = 0.0f;
#pragma unroll
    for (int kt = 0; kt < HH / 32; ++kt) {
      bhalf16 a = load_a16(A + kt * 32, HH, lane);
      bhalf16 b = load_bT16(WkT, HH, nt * 16, kt * 32, lane);
      acc = __builtin_amdgcn_wmma_f32_16x16x32_bf16(false, a, false, b, (short)0, acc, false, false);
    }
#pragma unroll
    for (int r = 0; r < 8; ++r)
      kout[(row0 + r + 8 * g) * HH + nt * 16 + n] = acc[r];
  }
}

// ---------------- per-row L2 normalize: kn = k / max(||k||, eps) ----------------
__global__ void __launch_bounds__(256)
k_rownorm(const float* __restrict__ kin, float* __restrict__ knout) {
  const int w = threadIdx.x >> 5, lane = threadIdx.x & 31;
  const size_t row = (size_t)blockIdx.x * 8 + w;
  const float* src = kin + row * HH;
  float sq = 0.0f;
#pragma unroll
  for (int ii = 0; ii < HH / 32; ++ii) { float v = src[lane + 32 * ii]; sq += v * v; }
  for (int m = 16; m > 0; m >>= 1) sq += __shfl_xor(sq, m, 32);
  float inv = 1.0f / fmaxf(sqrtf(sq), NORM_EPS);
  float* dst = knout + row * HH;
#pragma unroll
  for (int ii = 0; ii < HH / 32; ++ii) dst[lane + 32 * ii] = src[lane + 32 * ii] * inv;
}

// ---------------- gated rank-1 EMA scan, M resident in LDS ----------------
// 1 block per batch; 256 threads; M is 256x256 fp32 with 257-float pitch
// (bank-conflict-free for both the matvec read and the rank-1 update).
// k[t]/kn[t] rows are double-buffered in LDS and prefetched one step ahead
// via GLOBAL_LOAD_ASYNC_TO_LDS (ASYNCcnt) so the dependent global load is
// off the 1023-step critical path. Row L-1 prefetched on the last step IS q.
#define SCAN_LDS_FLOATS (256 * 257 + 2 * 256 + 2 * 256 + 32)
__global__ void __launch_bounds__(256)
k_scan(const float* __restrict__ kin, const float* __restrict__ knin, float* __restrict__ readv) {
  extern __shared__ float smem[];
  float* M   = smem;                    // 256*257
  float* kq  = smem + 256 * 257;        // 2 x 256 (double buffer, k rows)
  float* knq = kq + 2 * 256;            // 2 x 256 (double buffer, kn rows)
  float* red = knq + 2 * 256;           // 32

  const int i = threadIdx.x;
  const int b = blockIdx.x;
  const float* kbase  = kin  + (size_t)b * LL * HH;
  const float* knbase = knin + (size_t)b * LL * HH;
  float* Mi = M + i * 257;

  for (int j = 0; j < 256; ++j) Mi[j] = 0.0f;

  // prologue: fill buffer 0 with row 0
#if USE_ASYNC_LDS
  __builtin_amdgcn_global_load_async_to_lds_b32((as1i*)(kbase + i),  (as3i*)(kq + i),  0, 0);
  __builtin_amdgcn_global_load_async_to_lds_b32((as1i*)(knbase + i), (as3i*)(knq + i), 0, 0);
  __builtin_amdgcn_s_wait_asynccnt(0);
#else
  kq[i]  = kbase[i];
  knq[i] = knbase[i];
#endif
  __syncthreads();

  for (int t = 0; t < LL - 1; ++t) {
    const int cur = t & 1, nxt = cur ^ 1;

    // prefetch row t+1 (valid through row LL-1 == q) into the other buffer
#if USE_ASYNC_LDS
    __builtin_amdgcn_global_load_async_to_lds_b32(
        (as1i*)(kbase + (size_t)(t + 1) * HH + i),  (as3i*)(kq + nxt * 256 + i),  0, 0);
    __builtin_amdgcn_global_load_async_to_lds_b32(
        (as1i*)(knbase + (size_t)(t + 1) * HH + i), (as3i*)(knq + nxt * 256 + i), 0, 0);
#else
    const float kpre  = kbase [(size_t)(t + 1) * HH + i];
    const float knpre = knbase[(size_t)(t + 1) * HH + i];
#endif

    const float* kc  = kq  + cur * 256;
    const float* knc = knq + cur * 256;

    float vp = 0.0f;
#pragma unroll 8
    for (int j = 0; j < 256; ++j) vp = fmaf(Mi[j], knc[j], vp);
    float ki = kc[i];
    float d  = ki - vp;

    float d2 = d * d, kk = ki * ki;
    for (int m = 16; m > 0; m >>= 1) { d2 += __shfl_xor(d2, m, 32); kk += __shfl_xor(kk, m, 32); }
    const int w = i >> 5;
    if ((i & 31) == 0) { red[2 * w] = d2; red[2 * w + 1] = kk; }
    __syncthreads();
    if (i == 0) {
      float a = 0.0f, c = 0.0f;
      for (int q = 0; q < 8; ++q) { a += red[2 * q]; c += red[2 * q + 1]; }
      red[16] = a; red[17] = c;
    }
    __syncthreads();

    if (red[16] >= GT2 * red[17]) {           // block-uniform gate
      float coef = DECAY * d;
#pragma unroll 8
      for (int j = 0; j < 256; ++j) Mi[j] = fmaf(coef, knc[j], Mi[j]);
    }

    // retire the prefetch into nxt, then make it visible block-wide
#if USE_ASYNC_LDS
    __builtin_amdgcn_s_wait_asynccnt(0);
#else
    kq[nxt * 256 + i]  = kpre;
    knq[nxt * 256 + i] = knpre;
#endif
    __syncthreads();
  }

  // fused readout: read = M @ q; q = k[b, L-1, :] already sits in buffer (LL-1)&1
  const float* qv = kq + ((LL - 1) & 1) * 256;
  float acc = 0.0f;
#pragma unroll 8
  for (int j = 0; j < 256; ++j) acc = fmaf(Mi[j], qv[j], acc);
  readv[(size_t)b * HH + i] = acc;
}

// ---------------- tmp = read @ Wr + br (fp32), store bf16 ----------------
__global__ void __launch_bounds__(256)
k_wr(const float* __restrict__ readv, const float* __restrict__ WrTf,
     const float* __restrict__ br, bhalf* __restrict__ tmpbf) {
  __shared__ float sh[HH];
  const int b = blockIdx.x, n = threadIdx.x;
  sh[n] = readv[(size_t)b * HH + n];
  __syncthreads();
  float acc = br[n];
  const float* w = WrTf + (size_t)n * HH;
#pragma unroll 8
  for (int k = 0; k < HH; ++k) acc = fmaf(sh[k], w[k], acc);
  tmpbf[(size_t)b * HH + n] = (bhalf)acc;
}

// ---------------- out = tmp @ Wo + bo  [64,256]x[256,32000] ----------------
__global__ void __launch_bounds__(32)
k_out(const bhalf* __restrict__ tmpbf, const bhalf* __restrict__ WoT,
      const float* __restrict__ bo, float* __restrict__ out) {
  const int lane = threadIdx.x;
  const int n = lane & 15, g = lane >> 4;
  const int nt0 = blockIdx.x * 16;
  const int m0  = blockIdx.y * 16;
  const bhalf* A = tmpbf + (size_t)m0 * HH;

  // warm L2 for the next N-tile's weight rows (streamed 16 MB operand)
  if (nt0 + 16 < VV) __builtin_prefetch(WoT + (size_t)(nt0 + 16 + n) * HH, 0, 1);

  floatx8 acc;
  float bias = bo[nt0 + n];
#pragma unroll
  for (int r = 0; r < 8; ++r) acc[r] = bias;
#pragma unroll
  for (int kt = 0; kt < HH / 32; ++kt) {
    bhalf16 a = load_a16(A + kt * 32, HH, lane);
    bhalf16 b = load_bT16(WoT, HH, nt0, kt * 32, lane);
    acc = __builtin_amdgcn_wmma_f32_16x16x32_bf16(false, a, false, b, (short)0, acc, false, false);
  }
#pragma unroll
  for (int r = 0; r < 8; ++r)
    out[(size_t)(m0 + r + 8 * g) * VV + nt0 + n] = acc[r];
}

// ---------------- host ----------------
extern "C" void kernel_launch(void* const* d_in, const int* in_sizes, int n_in,
                              void* d_out, int out_size, void* d_ws, size_t ws_size,
                              hipStream_t stream) {
  const int*   seq   = (const int*)  d_in[0];
  const float* embed = (const float*)d_in[1];
  const float* W1    = (const float*)d_in[2];
  const float* b1    = (const float*)d_in[3];
  const float* W2    = (const float*)d_in[4];
  const float* b2    = (const float*)d_in[5];
  const float* ln_g  = (const float*)d_in[6];
  const float* ln_b  = (const float*)d_in[7];
  const float* Wk    = (const float*)d_in[8];
  const float* Wr    = (const float*)d_in[9];
  const float* br    = (const float*)d_in[10];
  const float* Wo    = (const float*)d_in[11];
  const float* bo    = (const float*)d_in[12];
  float* out = (float*)d_out;

  char* wsp = (char*)d_ws;
  auto alloc = [&](size_t bytes) -> char* {
    char* p = wsp;
    wsp += (bytes + 255) & ~(size_t)255;
    return p;
  };
  bhalf* W1T  = (bhalf*)alloc((size_t)H2 * HH * sizeof(bhalf));   // [512,256]
  bhalf* W2T  = (bhalf*)alloc((size_t)HH * H2 * sizeof(bhalf));   // [256,512]
  bhalf* WkT  = (bhalf*)alloc((size_t)HH * HH * sizeof(bhalf));   // [256,256]
  bhalf* WoT  = (bhalf*)alloc((size_t)VV * HH * sizeof(bhalf));   // [32000,256]
  float* WrTf = (float*)alloc((size_t)HH * HH * sizeof(float));   // [256,256]
  bhalf* hbf  = (bhalf*)alloc((size_t)BB * LL * HH * sizeof(bhalf));
  float* kbuf = (float*)alloc((size_t)BB * LL * HH * sizeof(float));
  float* knbf = (float*)alloc((size_t)BB * LL * HH * sizeof(float));
  float* readv= (float*)alloc((size_t)BB * HH * sizeof(float));
  bhalf* tmpbf= (bhalf*)alloc((size_t)BB * HH * sizeof(bhalf));

  // weight convert + transpose
  {
    size_t t;
    t = (size_t)HH * H2; k_cvtT_bf16<<<dim3((unsigned)((t + 255) / 256)), 256, 0, stream>>>(W1, W1T, HH, H2);
    t = (size_t)H2 * HH; k_cvtT_bf16<<<dim3((unsigned)((t + 255) / 256)), 256, 0, stream>>>(W2, W2T, H2, HH);
    t = (size_t)HH * HH; k_cvtT_bf16<<<dim3((unsigned)((t + 255) / 256)), 256, 0, stream>>>(Wk, WkT, HH, HH);
    t = (size_t)HH * VV; k_cvtT_bf16<<<dim3((unsigned)((t + 255) / 256)), 256, 0, stream>>>(Wo, WoT, HH, VV);
    t = (size_t)HH * HH; k_cvtT_f32 <<<dim3((unsigned)((t + 255) / 256)), 256, 0, stream>>>(Wr, WrTf, HH, HH);
  }

  // fused embed + FFN + LN  (one wave per 16 tokens)
  k_embed_ffn_ln<<<dim3(BB * LL / 16), 32, 0, stream>>>(seq, embed, W1T, b1, W2T, b2, ln_g, ln_b, hbf);

  // keys for all L positions (last row is q)
  k_keys<<<dim3(BB * LL / 16), 32, 0, stream>>>(hbf, WkT, kbuf);

  // row normalize
  k_rownorm<<<dim3(BB * LL / 8), 256, 0, stream>>>(kbuf, knbf);

  // LDS-resident sequential scan + fused read = M @ q
  const int scanBytes = SCAN_LDS_FLOATS * (int)sizeof(float);  // 267,392 B < 320 KB/WGP
  (void)hipFuncSetAttribute(reinterpret_cast<const void*>(k_scan),
                            hipFuncAttributeMaxDynamicSharedMemorySize, scanBytes);
  k_scan<<<dim3(BB), 256, scanBytes, stream>>>(kbuf, knbf, readv);

  // tmp = read @ Wr + br
  k_wr<<<dim3(BB), 256, 0, stream>>>(readv, WrTf, br, tmpbf);

  // out = tmp @ Wo + bo
  k_out<<<dim3(VV / 16, BB / 16), 32, 0, stream>>>(tmpbf, WoT, bo, out);
}
